// SNN_24292335026511
// MI455X (gfx1250) — matrix operand order repaired and checked by reading, not compiled, not run on hardware
//
#include <hip/hip_runtime.h>
#include <hip/hip_bf16.h>

// ---------------------------------------------------------------------------
// SNN forward: 2x recurrent LIF layers + leaky readout.
//   GEMMs in bf16 via v_wmma_f32_16x16x32_bf16. Spikes are exactly {0,1} in
//   bf16 so layers 2/3 are exact; only x/W quantize.
//   B operand (W^T tiles, column-major in memory) loaded with the CDNA5
//   plain GLOBAL_LOAD_B128 (clause of 4, immediate
//   offsets) -> no LDS, no barriers, no per-load address arithmetic.
//   GEMM dims are compile-time template constants so strides become
//   immediate offsets.
// ---------------------------------------------------------------------------

typedef __bf16 bf16_t;
typedef __bf16 v16bf __attribute__((ext_vector_type(16)));
typedef __bf16 v8bf  __attribute__((ext_vector_type(8)));
typedef float  v8f   __attribute__((ext_vector_type(8)));
typedef unsigned int v4u __attribute__((ext_vector_type(4)));

#define ALPHA 0.8187307530779818f   // exp(-dt/tau_syn)
#define BETA  0.9048374180359595f   // exp(-dt/tau_mem)

union Bfrag {                 // 32x16 B operand = two plain-loaded 16x16 tiles
  struct { v4u lo, hi; } u;
  v16bf v;
};
union Afrag {                 // 16x32 A operand
  struct { v8bf lo, hi; } h;
  v16bf v;
};

// ---------------------------------------------------------------------------
// fp32 [R,C] -> bf16 [Rp,Cp], zero padded. 2D grid: y = row, x*256+tid = col.
// ---------------------------------------------------------------------------
__global__ void snn_pad_convert_bf16(const float* __restrict__ src,
                                     bf16_t* __restrict__ dst,
                                     int R, int C, int Cp) {
  int c = blockIdx.x * blockDim.x + threadIdx.x;
  int r = blockIdx.y;
  if (c >= Cp) return;
  float v = (r < R && c < C) ? src[(long)r * C + c] : 0.0f;
  dst[(long)r * Cp + c] = (bf16_t)v;
}

// ---------------------------------------------------------------------------
// C[M,N] (f32) = A[M,K] (bf16 rm) * W[N,K]^T (bf16 rm)
// Block: 256 threads = 8 waves as 2(M) x 4(N). Wave tile: 64M x 32N
// (4 M-subtiles x 2 N-subtiles = 8 f32 accumulators).
// Workgroup tile: 128M x 128N. K step 32, fully unrolled.
// Per K-step per wave: 8x global_load_b128 (A) + 4x global_load_b128 (B,
// one s_clause, immediate offsets) + 8x v_wmma_f32_16x16x32_bf16.
// ---------------------------------------------------------------------------
template <int K, int N>
__global__ __launch_bounds__(256)
void snn_gemm_bf16_wmma(const bf16_t* __restrict__ A,
                        const bf16_t* __restrict__ W,
                        float* __restrict__ C) {
  const int tid    = threadIdx.x;
  const int lane   = tid & 31;
  const int lane15 = lane & 15;
  const int half   = lane >> 4;      // 0/1
  const int wid    = tid >> 5;
  const int mg     = wid & 1;        // 0..1 -> M group
  const int ng     = wid >> 1;       // 0..3 -> N group

  const int mBase = blockIdx.x * 128 + mg * 64;
  const int nBase = blockIdx.y * 128 + ng * 32;

  v8f acc[4][2] = {};

  // per-lane row pointers (A layout: lane15 = M row, halves split K)
  const bf16_t* arow[4];
#pragma unroll
  for (int s = 0; s < 4; ++s)
    arow[s] = A + (long)(mBase + s * 16 + lane15) * K + half * 8;
  // B tile per-lane row pointers (lane15 = N row of W, halves split K);
  // loop-invariant: K offsets folded into the load's immediate offset field.
  const unsigned long long wrow0 =
      (unsigned long long)(const void*)(W + (long)(nBase + lane15) * K + half * 8);
  const unsigned long long wrow1 =
      (unsigned long long)(const void*)(W + (long)(nBase + 16 + lane15) * K + half * 8);

#pragma unroll
  for (int k0 = 0; k0 < K; k0 += 32) {
    // ---- B fragments: clause of 4 plain 16x16 tile loads (lane = n, 8 contiguous k) ----
    Bfrag b0, b1;
    asm volatile(
        "s_clause 0x3\n\t"
        "global_load_b128 %0, %4, off offset:%6\n\t"
        "global_load_b128 %1, %4, off offset:%7\n\t"
        "global_load_b128 %2, %5, off offset:%6\n\t"
        "global_load_b128 %3, %5, off offset:%7"
        : "=v"(b0.u.lo), "=v"(b0.u.hi), "=v"(b1.u.lo), "=v"(b1.u.hi)
        : "v"(wrow0), "v"(wrow1),
          "i"(k0 * 2), "i"((k0 + 16) * 2));
    // fence: tie the load results so WMMAs can't move above the wait
    asm volatile("s_wait_loadcnt 0x0"
                 : "+v"(b0.u.lo), "+v"(b0.u.hi), "+v"(b1.u.lo), "+v"(b1.u.hi));

    // ---- 4 M sub-tiles x 2 N sub-tiles ----
#pragma unroll
    for (int s = 0; s < 4; ++s) {
      Afrag af;
      af.h.lo = *(const v8bf*)(arow[s] + k0);        // K = k0 + half*8 ..
      af.h.hi = *(const v8bf*)(arow[s] + k0 + 16);   // K = k0+16+half*8 ..
      acc[s][0] = __builtin_amdgcn_wmma_f32_16x16x32_bf16(
                      false, af.v, false, b0.v, (short)0, acc[s][0], false, false);
      acc[s][1] = __builtin_amdgcn_wmma_f32_16x16x32_bf16(
                      false, af.v, false, b1.v, (short)0, acc[s][1], false, false);
    }
  }

  // ---- store: accumulator VGPR r, lane -> (m = base + r + half*8, n = lane15)
#pragma unroll
  for (int s = 0; s < 4; ++s) {
#pragma unroll
    for (int t = 0; t < 2; ++t) {
      float* p = C + (long)(mBase + s * 16 + half * 8) * N + nBase + t * 16 + lane15;
#pragma unroll
      for (int r = 0; r < 8; ++r)
        p[r * N] = acc[s][t][r];   // N compile-time -> immediate offsets
    }
  }
}

// ---------------------------------------------------------------------------
// Recurrent LIF scan. h layout: [B*T, O], row = b*T + t.
// Spike emitted from pre-update membrane (matches lax.scan ordering).
// ---------------------------------------------------------------------------
__global__ void snn_scan_recurrent(const float* __restrict__ h,
                                   bf16_t* __restrict__ spk,
                                   int B, int T, int O) {
  int idx = blockIdx.x * blockDim.x + threadIdx.x;   // b*O + o
  if (idx >= B * O) return;
  int b = idx / O, o = idx % O;
  const float* hp = h   + (long)b * T * O + o;
  bf16_t*      sp = spk + (long)b * T * O + o;
  float syn = 0.0f, mem = 0.0f;
  for (int t = 0; t < T; ++t) {
    float out = (mem > 1.0f) ? 1.0f : 0.0f;    // spike(mem - 1)
    sp[(long)t * O] = (bf16_t)out;
    float ht   = hp[(long)t * O];
    float nsyn = ALPHA * syn + ht;
    float nmem = (BETA * mem + syn) * (1.0f - out);
    syn = nsyn;
    mem = nmem;
  }
}

// ---------------------------------------------------------------------------
// Readout scan: new_flt = a*flt + h; new_out = b*out + flt_old; emit new_out.
// ---------------------------------------------------------------------------
__global__ void snn_scan_readout(const float* __restrict__ h2,
                                 float* __restrict__ out,
                                 int B, int T, int O, int Op) {
  int idx = blockIdx.x * blockDim.x + threadIdx.x;   // b*O + o
  if (idx >= B * O) return;
  int b = idx / O, o = idx % O;
  const float* hp = h2  + (long)b * T * Op + o;
  float*       op = out + (long)b * T * O  + o;
  float flt = 0.0f, mo = 0.0f;
  for (int t = 0; t < T; ++t) {
    float ht = hp[(long)t * Op];
    float nf = ALPHA * flt + ht;
    float no = BETA * mo + flt;       // uses pre-update flt
    op[(long)t * O] = no;
    flt = nf;
    mo  = no;
  }
}

// ---------------------------------------------------------------------------
// launch
// ---------------------------------------------------------------------------
extern "C" void kernel_launch(void* const* d_in, const int* in_sizes, int n_in,
                              void* d_out, int out_size, void* d_ws, size_t ws_size,
                              hipStream_t stream) {
  (void)in_sizes; (void)n_in; (void)out_size; (void)ws_size;

  const int B = 128, T = 500, NIN = 700, KP0 = 704, H = 512;
  const int NOUT = 20, NOUTP = 128;
  const int M = B * T;                         // 64000

  const float* x  = (const float*)d_in[0];     // [B,T,NIN]
  const float* W0 = (const float*)d_in[1];     // [H,NIN]
  const float* W1 = (const float*)d_in[2];     // [H,H]
  const float* W2 = (const float*)d_in[3];     // [NOUT,H]
  float* outp = (float*)d_out;                 // [B,T,NOUT]

  char* ws = (char*)d_ws;
  bf16_t* xbf  = (bf16_t*)(ws);                 // 64000*704 bf16 = 90,112,000
  bf16_t* w0b  = (bf16_t*)(ws +  90112000);     // 512*704   bf16 =    720,896
  bf16_t* w1b  = (bf16_t*)(ws +  90832896);     // 512*512   bf16 =    524,288
  bf16_t* w2b  = (bf16_t*)(ws +  91357184);     // 128*512   bf16 =    131,072
  float*  hbuf = (float*) (ws +  91488256);     // 64000*512 f32  = 131,072,000 (h0, reused h1)
  bf16_t* s1   = (bf16_t*)(ws + 222560256);     // 64000*512 bf16 =  65,536,000
  bf16_t* s2   = (bf16_t*)(ws + 288096256);     // 64000*512 bf16 =  65,536,000
  float*  h2   = (float*) (ws + 353632256);     // 64000*128 f32  =  32,768,000
  // total ws usage: 386,400,256 bytes

  // --- convert / pad inputs to bf16 (2D grid: y = row) ---
  snn_pad_convert_bf16<<<dim3((KP0 + 255) / 256, M), 256, 0, stream>>>(x, xbf, M, NIN, KP0);
  snn_pad_convert_bf16<<<dim3((KP0 + 255) / 256, H), 256, 0, stream>>>(W0, w0b, H, NIN, KP0);
  snn_pad_convert_bf16<<<dim3((H + 255) / 256, H), 256, 0, stream>>>(W1, w1b, H, H, H);
  snn_pad_convert_bf16<<<dim3((H + 255) / 256, NOUTP), 256, 0, stream>>>(W2, w2b, NOUT, H, H);

  // --- layer 1: GEMM + scan ---
  snn_gemm_bf16_wmma<KP0, H><<<dim3(M / 128, H / 128), 256, 0, stream>>>(xbf, w0b, hbuf);
  snn_scan_recurrent<<<(B * H + 255) / 256, 256, 0, stream>>>(hbuf, s1, B, T, H);

  // --- layer 2: GEMM + scan ---
  snn_gemm_bf16_wmma<H, H><<<dim3(M / 128, H / 128), 256, 0, stream>>>(s1, w1b, hbuf);
  snn_scan_recurrent<<<(B * H + 255) / 256, 256, 0, stream>>>(hbuf, s2, B, T, H);

  // --- readout: GEMM (N padded to 128) + scan ---
  snn_gemm_bf16_wmma<H, NOUTP><<<dim3(M / 128, NOUTP / 128), 256, 0, stream>>>(s2, w2b, h2);
  snn_scan_readout<<<(B * NOUT + 255) / 256, 256, 0, stream>>>(h2, outp, B, T, NOUT, NOUTP);
}
